// StyleLossW2_2233382994377
// MI455X (gfx1250) — matrix operand order repaired and checked by reading, not compiled, not run on hardware
//
#include <hip/hip_runtime.h>
#include <math.h>

// ---------------------------------------------------------------------------
// StyleLossW2: x[8,512,128,128] f32, target_mean[512], target_srm[512,512]
// out: scalar f32.
// Heavy work = 512x512 matmuls -> v_wmma_f32_16x16x32_bf16 with f32 accum.
// 4-wave workgroups, 32x32 per-wave register tiles (4 accumulators) to
// maximize WMMA per LDS fragment load.
// ---------------------------------------------------------------------------

#define NB     8
#define C512   512
#define HW     16384          // 128*128
#define MAT    (512 * 512)    // 262144
#define EPSV   1e-4f
#define NS_IT  12

typedef __attribute__((ext_vector_type(16))) __bf16 bf16x16;
typedef __attribute__((ext_vector_type(8)))  __bf16 bf16x8;
typedef __attribute__((ext_vector_type(8)))  float  v8f;

__device__ __forceinline__ bf16x16 cat8(bf16x8 lo, bf16x8 hi) {
    return __builtin_shufflevector(lo, hi, 0,1,2,3,4,5,6,7,8,9,10,11,12,13,14,15);
}

// ---------------------------------------------------------------------------
// Per-(b,c) spatial mean over HW elements.
// ---------------------------------------------------------------------------
__global__ void __launch_bounds__(256)
mean_kernel(const float* __restrict__ x, float* __restrict__ mean) {
    __shared__ float red[256];
    const int idx = blockIdx.x;                 // b*512 + c
    const float* p = x + (size_t)idx * HW;
    float s = 0.f;
    for (int i = threadIdx.x; i < HW; i += 256) s += p[i];
    red[threadIdx.x] = s;
    __syncthreads();
    for (int off = 128; off > 0; off >>= 1) {
        if (threadIdx.x < off) red[threadIdx.x] += red[threadIdx.x + off];
        __syncthreads();
    }
    if (threadIdx.x == 0) mean[idx] = red[0] * (1.f / (float)HW);
}

// ---------------------------------------------------------------------------
// tcov = target_srm - outer(tmean,tmean) + EPS*I
// ---------------------------------------------------------------------------
__global__ void __launch_bounds__(256)
tcov_kernel(const float* __restrict__ tsrm, const float* __restrict__ tmean,
            float* __restrict__ tcov) {
    const int e = blockIdx.x * 256 + threadIdx.x;   // < 262144
    const int i = e >> 9, j = e & 511;
    tcov[e] = tsrm[e] - tmean[i] * tmean[j] + ((i == j) ? EPSV : 0.f);
}

// ---------------------------------------------------------------------------
// Wave-level 32x32 WMMA micro-tile: loads 2 A-frags + 2 B-frags from LDS,
// performs 4 wmma accumulations.
// ---------------------------------------------------------------------------
__device__ __forceinline__ void wave_tile_32x32(
    const __bf16 (*As)[40], const __bf16 (*Bs)[40],
    int R, int Cb, int lane,
    v8f& a00, v8f& a01, v8f& a10, v8f& a11) {
    const int r   = lane & 15;
    const int kbA = (lane >> 4) * 8;     // A: K 0-7/16-23 vs 8-15/24-31
    bf16x16 fa0 = cat8(*(const bf16x8*)&As[R + r][kbA],
                       *(const bf16x8*)&As[R + r][kbA + 16]);
    bf16x16 fa1 = cat8(*(const bf16x8*)&As[R + 16 + r][kbA],
                       *(const bf16x8*)&As[R + 16 + r][kbA + 16]);
    const int kbB = (lane >> 4) * 8;     // B: K 0-7/16-23 vs 8-15/24-31
    bf16x16 fb0 = cat8(*(const bf16x8*)&Bs[Cb + r][kbB],
                       *(const bf16x8*)&Bs[Cb + r][kbB + 16]);
    bf16x16 fb1 = cat8(*(const bf16x8*)&Bs[Cb + 16 + r][kbB],
                       *(const bf16x8*)&Bs[Cb + 16 + r][kbB + 16]);
    a00 = __builtin_amdgcn_wmma_f32_16x16x32_bf16(false, fa0, false, fb0, (short)0, a00, false, false);
    a01 = __builtin_amdgcn_wmma_f32_16x16x32_bf16(false, fa0, false, fb1, (short)0, a01, false, false);
    a10 = __builtin_amdgcn_wmma_f32_16x16x32_bf16(false, fa1, false, fb0, (short)0, a10, false, false);
    a11 = __builtin_amdgcn_wmma_f32_16x16x32_bf16(false, fa1, false, fb1, (short)0, a11, false, false);
}

// ---------------------------------------------------------------------------
// cov[b] = X Xt / HW - outer(mean_b,mean_b) + EPS*I  via bf16 WMMA.
// Grid (8 ntile, 8 mtile, 8 batch), 128 threads (4 waves), 64x64 WG tile,
// each wave owns a 32x32 sub-tile.
// ---------------------------------------------------------------------------
__global__ void __launch_bounds__(128)
gram_cov_kernel(const float* __restrict__ x, const float* __restrict__ mean,
                float* __restrict__ cov) {
    __shared__ __bf16 As[64][40];
    __shared__ __bf16 Bs[64][40];

    const int b  = blockIdx.z;
    const float* X = x + (size_t)b * C512 * HW;
    const int m0 = blockIdx.y * 64;
    const int n0 = blockIdx.x * 64;
    const int tid  = threadIdx.x;
    const int lane = tid & 31;
    const int w    = tid >> 5;
    const int R    = (w >> 1) * 32;   // wave rows within tile
    const int Cb   = (w & 1) * 32;    // wave cols within tile

    v8f a00 = {}, a01 = {}, a10 = {}, a11 = {};

    for (int k0 = 0; k0 < HW; k0 += 32) {
        // stage 64x32 f32 tiles -> bf16 LDS (rows of X are contiguous)
        #pragma unroll
        for (int i = 0; i < 4; ++i) {
            int e  = tid + i * 128;
            int r  = e >> 3;
            int c4 = (e & 7) * 4;
            float4 va = *(const float4*)(X + (size_t)(m0 + r) * HW + k0 + c4);
            As[r][c4 + 0] = (__bf16)va.x; As[r][c4 + 1] = (__bf16)va.y;
            As[r][c4 + 2] = (__bf16)va.z; As[r][c4 + 3] = (__bf16)va.w;
            float4 vb = *(const float4*)(X + (size_t)(n0 + r) * HW + k0 + c4);
            Bs[r][c4 + 0] = (__bf16)vb.x; Bs[r][c4 + 1] = (__bf16)vb.y;
            Bs[r][c4 + 2] = (__bf16)vb.z; Bs[r][c4 + 3] = (__bf16)vb.w;
        }
        __syncthreads();
        wave_tile_32x32(As, Bs, R, Cb, lane, a00, a01, a10, a11);
        __syncthreads();
    }

    const float invhw = 1.f / (float)HW;
    const float* mb = mean + b * C512;
    const int n  = lane & 15;
    const int mo = (lane >> 4) * 8;
    float* D = cov + (size_t)b * MAT;
    v8f accs[2][2] = {{a00, a01}, {a10, a11}};
    #pragma unroll
    for (int ar = 0; ar < 2; ++ar) {
        #pragma unroll
        for (int bc = 0; bc < 2; ++bc) {
            const int gmb = m0 + R + ar * 16 + mo;
            const int gn  = n0 + Cb + bc * 16 + n;
            #pragma unroll
            for (int v = 0; v < 8; ++v) {
                int rm = gmb + v;
                float d = accs[ar][bc][v] * invhw - mb[rm] * mb[gn]
                        + ((rm == gn) ? EPSV : 0.f);
                D[(size_t)rm * C512 + gn] = d;
            }
        }
    }
}

// ---------------------------------------------------------------------------
// Batched D[b] = alpha * A[b] @ B[b] + beta_diag * I   (512x512 row-major)
// Strides in floats; 0 broadcasts. Grid (8,8,nb), 128 threads (4 waves).
// ---------------------------------------------------------------------------
__global__ void __launch_bounds__(128)
mm512_kernel(const float* __restrict__ A, const float* __restrict__ B,
             float* __restrict__ D, int sA, int sB, int sD,
             float alpha, float beta_diag) {
    __shared__ __bf16 As[64][40];
    __shared__ __bf16 Bs[64][40];   // Bs[n][k] = B[k][n]  (transposed)

    const int bz = blockIdx.z;
    A += (size_t)bz * sA;
    B += (size_t)bz * sB;
    D += (size_t)bz * sD;
    const int m0 = blockIdx.y * 64;
    const int n0 = blockIdx.x * 64;
    const int tid  = threadIdx.x;
    const int lane = tid & 31;
    const int w    = tid >> 5;
    const int R    = (w >> 1) * 32;
    const int Cb   = (w & 1) * 32;

    v8f a00 = {}, a01 = {}, a10 = {}, a11 = {};

    for (int k0 = 0; k0 < C512; k0 += 32) {
        #pragma unroll
        for (int i = 0; i < 4; ++i) {
            int e  = tid + i * 128;
            int r  = e >> 3;
            int c4 = (e & 7) * 4;
            float4 va = *(const float4*)(A + (size_t)(m0 + r) * C512 + k0 + c4);
            As[r][c4 + 0] = (__bf16)va.x; As[r][c4 + 1] = (__bf16)va.y;
            As[r][c4 + 2] = (__bf16)va.z; As[r][c4 + 3] = (__bf16)va.w;
        }
        #pragma unroll
        for (int i = 0; i < 16; ++i) {
            int e = tid + i * 128;      // 64 n x 32 k
            int k = e >> 6;
            int n = e & 63;
            Bs[n][k] = (__bf16)B[(size_t)(k0 + k) * C512 + n0 + n];
        }
        __syncthreads();
        wave_tile_32x32(As, Bs, R, Cb, lane, a00, a01, a10, a11);
        __syncthreads();
    }

    const int n  = lane & 15;
    const int mo = (lane >> 4) * 8;
    v8f accs[2][2] = {{a00, a01}, {a10, a11}};
    #pragma unroll
    for (int ar = 0; ar < 2; ++ar) {
        #pragma unroll
        for (int bc = 0; bc < 2; ++bc) {
            const int gmb = m0 + R + ar * 16 + mo;
            const int gn  = n0 + Cb + bc * 16 + n;
            #pragma unroll
            for (int v = 0; v < 8; ++v) {
                int rm = gmb + v;
                float d = alpha * accs[ar][bc][v] + ((rm == gn) ? beta_diag : 0.f);
                D[(size_t)rm * C512 + gn] = d;
            }
        }
    }
}

// ---------------------------------------------------------------------------
// Frobenius norm per batch: norms[b] = sqrt(sum A[b]^2). One WG per batch.
// ---------------------------------------------------------------------------
__global__ void __launch_bounds__(256)
fro_norm_kernel(const float* __restrict__ A, int stride, float* __restrict__ norms) {
    __shared__ float red[256];
    const float* p = A + (size_t)blockIdx.x * stride;
    float s = 0.f;
    for (int i = threadIdx.x; i < MAT; i += 256) { float v = p[i]; s += v * v; }
    red[threadIdx.x] = s;
    __syncthreads();
    for (int off = 128; off > 0; off >>= 1) {
        if (threadIdx.x < off) red[threadIdx.x] += red[threadIdx.x + off];
        __syncthreads();
    }
    if (threadIdx.x == 0) norms[blockIdx.x] = sqrtf(red[0]);
}

// Y[b] = A[b]/norms[b]; Z[b] = I.  Grid (1024, nb).
__global__ void __launch_bounds__(256)
ns_init_kernel(const float* __restrict__ A, const float* __restrict__ norms,
               float* __restrict__ Y, float* __restrict__ Z) {
    const int b = blockIdx.y;
    const int e = blockIdx.x * 256 + threadIdx.x;
    const size_t o = (size_t)b * MAT + e;
    const float inv = 1.f / norms[b];
    Y[o] = A[o] * inv;
    const int i = e >> 9, j = e & 511;
    Z[o] = (i == j) ? 1.f : 0.f;
}

// tcov_sqrt = Y * sqrt(norms[0])
__global__ void __launch_bounds__(256)
scale_sqrt_kernel(const float* __restrict__ Y, const float* __restrict__ norms,
                  float* __restrict__ dst) {
    const int e = blockIdx.x * 256 + threadIdx.x;
    dst[e] = Y[e] * sqrtf(norms[0]);
}

// ---------------------------------------------------------------------------
// out = mean((mean-tmean)^2) + mean_diag(tcov + cov - 2*Yf*sqrt(norm))
// ---------------------------------------------------------------------------
__global__ void __launch_bounds__(256)
final_kernel(const float* __restrict__ mean, const float* __restrict__ tmean,
             const float* __restrict__ tcov, const float* __restrict__ cov,
             const float* __restrict__ Yf, const float* __restrict__ normsB,
             float* __restrict__ out) {
    __shared__ float r1[256];
    __shared__ float r2[256];
    float s1 = 0.f, s2 = 0.f;
    for (int e = threadIdx.x; e < NB * C512; e += 256) {
        const int b = e >> 9, c = e & 511;
        float d = mean[e] - tmean[c];
        s1 += d * d;
        float sq = sqrtf(normsB[b]);
        const int di = c * 513;
        s2 += tcov[di] + cov[(size_t)b * MAT + di] - 2.f * Yf[(size_t)b * MAT + di] * sq;
    }
    r1[threadIdx.x] = s1; r2[threadIdx.x] = s2;
    __syncthreads();
    for (int off = 128; off > 0; off >>= 1) {
        if (threadIdx.x < off) {
            r1[threadIdx.x] += r1[threadIdx.x + off];
            r2[threadIdx.x] += r2[threadIdx.x + off];
        }
        __syncthreads();
    }
    if (threadIdx.x == 0) out[0] = (r1[0] + r2[0]) * (1.f / (float)(NB * C512));
}

// ---------------------------------------------------------------------------
extern "C" void kernel_launch(void* const* d_in, const int* in_sizes, int n_in,
                              void* d_out, int out_size, void* d_ws, size_t ws_size,
                              hipStream_t stream) {
    const float* x     = (const float*)d_in[0];
    const float* tmean = (const float*)d_in[1];
    const float* tsrm  = (const float*)d_in[2];
    float* out = (float*)d_out;
    float* ws  = (float*)d_ws;

    // workspace layout (floats): ~13.1M floats (~52.5 MB)
    float* mean   = ws;                      // 4096
    float* norms  = ws + 4096;               // 16 ([0]=target, [1..8]=batch)
    float* tcov   = norms + 16;              // MAT
    float* tcovs  = tcov + MAT;              // MAT
    float* cov    = tcovs + MAT;             // 8*MAT
    float* T      = cov  + (size_t)NB * MAT; // 8*MAT
    float* Y0     = T    + (size_t)NB * MAT; // 8*MAT
    float* Y1     = Y0   + (size_t)NB * MAT; // 8*MAT
    float* Z0     = Y1   + (size_t)NB * MAT; // 8*MAT
    float* Z1     = Z0   + (size_t)NB * MAT; // 8*MAT

    // ---- statistics ----
    mean_kernel<<<NB * C512, 256, 0, stream>>>(x, mean);
    tcov_kernel<<<MAT / 256, 256, 0, stream>>>(tsrm, tmean, tcov);
    gram_cov_kernel<<<dim3(8, 8, NB), 128, 0, stream>>>(x, mean, cov);

    // ---- Newton-Schulz sqrt of tcov (batch 1) ----
    fro_norm_kernel<<<1, 256, 0, stream>>>(tcov, 0, norms);
    ns_init_kernel<<<dim3(MAT / 256, 1), 256, 0, stream>>>(tcov, norms, Y0, Z0);
    {
        float *Yc = Y0, *Zc = Z0, *Yn = Y1, *Zn = Z1;
        for (int it = 0; it < NS_IT; ++it) {
            mm512_kernel<<<dim3(8, 8, 1), 128, 0, stream>>>(Zc, Yc, T, 0, 0, 0, -0.5f, 1.5f);
            mm512_kernel<<<dim3(8, 8, 1), 128, 0, stream>>>(Yc, T, Yn, 0, 0, 0, 1.f, 0.f);
            mm512_kernel<<<dim3(8, 8, 1), 128, 0, stream>>>(T, Zc, Zn, 0, 0, 0, 1.f, 0.f);
            float* t;
            t = Yc; Yc = Yn; Yn = t;
            t = Zc; Zc = Zn; Zn = t;
        }
        scale_sqrt_kernel<<<MAT / 256, 256, 0, stream>>>(Yc, norms, tcovs);
    }

    // ---- M[b] = tcov_sqrt @ cov[b] @ tcov_sqrt  (tmp in Z1, M in T) ----
    mm512_kernel<<<dim3(8, 8, NB), 128, 0, stream>>>(tcovs, cov, Z1, 0, MAT, MAT, 1.f, 0.f);
    mm512_kernel<<<dim3(8, 8, NB), 128, 0, stream>>>(Z1, tcovs, T, MAT, 0, MAT, 1.f, 0.f);

    // ---- batched Newton-Schulz sqrt of M (batch 8) ----
    fro_norm_kernel<<<NB, 256, 0, stream>>>(T, MAT, norms + 1);
    ns_init_kernel<<<dim3(MAT / 256, NB), 256, 0, stream>>>(T, norms + 1, Y0, Z0);
    float *Yc = Y0, *Zc = Z0, *Yn = Y1, *Zn = Z1;
    for (int it = 0; it < NS_IT; ++it) {
        mm512_kernel<<<dim3(8, 8, NB), 128, 0, stream>>>(Zc, Yc, T, MAT, MAT, MAT, -0.5f, 1.5f);
        mm512_kernel<<<dim3(8, 8, NB), 128, 0, stream>>>(Yc, T, Yn, MAT, MAT, MAT, 1.f, 0.f);
        mm512_kernel<<<dim3(8, 8, NB), 128, 0, stream>>>(T, Zc, Zn, MAT, MAT, MAT, 1.f, 0.f);
        float* t;
        t = Yc; Yc = Yn; Yn = t;
        t = Zc; Zc = Zn; Zn = t;
    }

    // ---- final scalar loss ----
    final_kernel<<<1, 256, 0, stream>>>(mean, tmean, tcov, cov, Yc, norms + 1, out);
}
